// CausalSelfAttention_77309411778
// MI455X (gfx1250) — matrix operand order repaired and checked by reading, not compiled, not run on hardware
//
#include <hip/hip_runtime.h>
#include <cstdint>

typedef __bf16 bf16_t;
typedef __attribute__((ext_vector_type(16))) __bf16 v16bf;
typedef __attribute__((ext_vector_type(8)))  float  v8f;
typedef __attribute__((ext_vector_type(4)))  unsigned ui4;
typedef __attribute__((ext_vector_type(8)))  int      i8v;
typedef __attribute__((ext_vector_type(4)))  int      i4v;

#define T_SEQ 4096
#define DMODEL 2048
#define NHEAD 16
#define HDHEAD 128

#if defined(__HIP_DEVICE_COMPILE__) && __has_builtin(__builtin_amdgcn_tensor_load_to_lds) && __has_builtin(__builtin_amdgcn_s_wait_tensorcnt)
#define HAVE_TDM 1
#else
#define HAVE_TDM 0
#endif

#if HAVE_TDM
// Tensor Data Mover: DMA a [rows x cols] bf16 tile (row stride in elements)
// from global memory into LDS (packed row-major, row length = cols).
// D# layout per CDNA5 ISA. Issued by one wave; completion
// via TENSORcnt (s_wait_tensorcnt 0) before the workgroup barrier.
__device__ __forceinline__ void tdm_load_bf16_2d(void* lds_dst, const void* gsrc,
                                                 unsigned cols, unsigned rows,
                                                 unsigned stride_elems) {
  unsigned long long ga = (unsigned long long)(uintptr_t)gsrc;
  ui4 g0;
  g0.x = 1u;                                            // count=1 (valid descriptor)
  g0.y = (unsigned)(uintptr_t)lds_dst;                  // LDS byte address (flat addr low 32 bits)
  g0.z = (unsigned)ga;                                  // global_addr[31:0]
  g0.w = (unsigned)((ga >> 32) & 0x01FFFFFFull) | (2u << 30);  // global_addr[56:32], type=2
  i8v g1;
  g1[0] = (int)(1u << 16);          // data_size=1 (2 bytes/elem)
  g1[1] = (int)(cols << 16);        // tensor_dim0[15:0]
  g1[2] = (int)(rows << 16);        // tensor_dim0[31:16]=0 | tensor_dim1[15:0]=rows
  g1[3] = (int)(cols << 16);        // tensor_dim1[31:16]=0 | tile_dim0=cols
  g1[4] = (int)rows;                // tile_dim1=rows | tile_dim2=0
  g1[5] = (int)stride_elems;        // tensor_dim0_stride[31:0]
  g1[6] = 0;
  g1[7] = 0;
  i4v gz = (i4v){0, 0, 0, 0};
#if __clang_major__ >= 23
  i8v gz8 = (i8v){0, 0, 0, 0, 0, 0, 0, 0};
  __builtin_amdgcn_tensor_load_to_lds(g0, g1, gz, gz, gz8, 0);
#else
  __builtin_amdgcn_tensor_load_to_lds(g0, g1, gz, gz, 0);
#endif
}
#endif

__device__ __forceinline__ v8f wmma_bf16(v16bf a, v16bf b, v8f c) {
  // D = A(16x32 bf16) * B(32x16 bf16) + C(16x16 f32)
  return __builtin_amdgcn_wmma_f32_16x16x32_bf16(false, a, false, b, (short)0, c, false, false);
}

// A-fragment (16x32, M x K): lane<16 -> row=lane, K = kk+{0..7,16..23}; lane>=16 -> K = kk+{8..15,24..31}
__device__ __forceinline__ v16bf load_afrag(const bf16_t* base, int stride, int kk, int lane) {
  int l16 = lane & 15, hi = lane >> 4;
  const bf16_t* p = base + l16 * stride + kk + hi * 8;
  v16bf a;
#pragma unroll
  for (int i = 0; i < 8; ++i) { a[i] = p[i]; a[i + 8] = p[i + 16]; }
  return a;
}

// B-fragment (32x16, K x N): lane<16 -> col=lane, K = kk+{0..7,16..23}; lane>=16 -> K = kk+{8..15,24..31}
__device__ __forceinline__ v16bf load_bfrag(const bf16_t* base, int stride, int kk, int lane) {
  int l16 = lane & 15, hi = lane >> 4;
  const bf16_t* p = base + l16 * stride + kk + hi * 8;
  v16bf b;
#pragma unroll
  for (int i = 0; i < 8; ++i) { b[i] = p[i]; b[i + 8] = p[i + 16]; }
  return b;
}

// ---------------- fp32 -> bf16 conversion of x ----------------
__global__ void __launch_bounds__(256) cvt_kernel(const float* __restrict__ in, bf16_t* __restrict__ out) {
  size_t i = ((size_t)blockIdx.x * 256 + threadIdx.x) * 4;
  float4 v = *(const float4*)(in + i);
  out[i + 0] = (bf16_t)v.x; out[i + 1] = (bf16_t)v.y;
  out[i + 2] = (bf16_t)v.z; out[i + 3] = (bf16_t)v.w;
}

// ---------------- shared GEMM core: C[64x128] += A[64xK](bf16) * W[128xK]^T(f32->bf16) ---------
__device__ __forceinline__ void gemm_tile(const bf16_t* __restrict__ A, int lda,
                                          const float* __restrict__ W, int ldw,
                                          int row0, int n0, int K,
                                          bf16_t (*At)[64], bf16_t (*Wt)[64],
                                          v8f acc[4]) {
  const int tid = threadIdx.x;
  const int lane = tid & 31, w = tid >> 5;
  const int rw = (w & 3) * 16, cw = (w >> 2) * 64;
#pragma unroll
  for (int t = 0; t < 4; ++t) acc[t] = (v8f){0.f,0.f,0.f,0.f,0.f,0.f,0.f,0.f};

  for (int k0 = 0; k0 < K; k0 += 64) {
    // A tile 64x64 bf16 -> LDS via Tensor Data Mover (wave 0 issues the DMA)
#if HAVE_TDM
    if (w == 0)
      tdm_load_bf16_2d(&At[0][0], A + (size_t)row0 * lda + k0, 64, 64, (unsigned)lda);
#else
#pragma unroll
    for (int it = 0; it < 2; ++it) {
      int c = tid + it * 256;
      int r = c >> 3, cc = (c & 7) * 8;
      *(uint4*)&At[r][cc] = *(const uint4*)(A + (size_t)(row0 + r) * lda + k0 + cc);
    }
#endif
    // W tile 128x64: f32 -> bf16 (float4 per chunk; 2048 chunks)
#pragma unroll
    for (int it = 0; it < 8; ++it) {
      int c = tid + it * 256;
      int r = c >> 4, cc = (c & 15) * 4;
      float4 v = *(const float4*)(W + (size_t)(n0 + r) * ldw + k0 + cc);
      Wt[r][cc + 0] = (bf16_t)v.x; Wt[r][cc + 1] = (bf16_t)v.y;
      Wt[r][cc + 2] = (bf16_t)v.z; Wt[r][cc + 3] = (bf16_t)v.w;
    }
    // warm GL2 for the next K step (global_prefetch_b8)
    if (k0 + 64 < K) {
      __builtin_prefetch(A + (size_t)(row0 + (tid & 63)) * lda + k0 + 64, 0, 1);
      __builtin_prefetch(W + (size_t)(n0 + (tid >> 1)) * ldw + k0 + 64, 0, 1);
    }
#if HAVE_TDM
    if (w == 0) __builtin_amdgcn_s_wait_tensorcnt(0);   // DMA visible before barrier
#endif
    __syncthreads();
#pragma unroll
    for (int kk = 0; kk < 64; kk += 32) {
      v16bf a = load_afrag(&At[rw][0], 64, kk, lane);
#pragma unroll
      for (int t = 0; t < 4; ++t) {
        v16bf b = load_bfrag(&Wt[cw + 16 * t][0], 64, kk, lane);
        acc[t] = wmma_bf16(a, b, acc[t]);
      }
    }
    __syncthreads();
  }
}

// ---------------- QKV projection + RMSNorm + RoPE + v-mix ----------------
__global__ void __launch_bounds__(256) qkv_kernel(const bf16_t* __restrict__ xb,
                                                  const float* __restrict__ qkv_w,
                                                  const float* __restrict__ ve,
                                                  const float* __restrict__ lambdas,
                                                  bf16_t* __restrict__ qb,
                                                  bf16_t* __restrict__ kb,
                                                  bf16_t* __restrict__ vb) {
  __shared__ bf16_t At[64][64];      // 8 KB
  __shared__ bf16_t Wt[128][64];     // 16 KB
  __shared__ float  CT[64][128];     // 32 KB
  __shared__ float  rfac[64];

  const int tid = threadIdx.x, lane = tid & 31, w = tid >> 5;
  const int l16 = lane & 15, hi = lane >> 4;
  const int row0 = blockIdx.x * 64;
  const int n0   = blockIdx.y * 128;   // one head per column block
  const int cc   = blockIdx.z;         // 0=q 1=k 2=v

  v8f acc[4];
  gemm_tile(xb, DMODEL, qkv_w + (size_t)cc * DMODEL * DMODEL, DMODEL, row0, n0, DMODEL, At, Wt, acc);

  // stage C tile to LDS (C layout: vgpr r -> row r + 8*hi, col = lane&15)
  const int rw = (w & 3) * 16, cw = (w >> 2) * 64;
#pragma unroll
  for (int t = 0; t < 4; ++t)
#pragma unroll
    for (int r = 0; r < 8; ++r)
      CT[rw + r + hi * 8][cw + 16 * t + l16] = acc[t][r];
  __syncthreads();

  if (cc == 2) {
    const float l0 = lambdas[0], l1 = lambdas[1];
    for (int e = tid; e < 64 * 128; e += 256) {
      int row = e >> 7, col = e & 127;
      int tg = row0 + row;
      float val = l0 * CT[row][col] + l1 * ve[(size_t)tg * DMODEL + n0 + col];
      vb[(size_t)tg * DMODEL + n0 + col] = (bf16_t)val;
    }
  } else {
    if (tid < 64) {
      float ss = 0.f;
      for (int c2 = 0; c2 < 128; ++c2) { float v = CT[tid][c2]; ss += v * v; }
      rfac[tid] = rsqrtf(ss * (1.0f / 128.0f) + 1e-6f);
    }
    __syncthreads();
    bf16_t* dst = (cc == 0) ? qb : kb;
    for (int e = tid; e < 64 * 128; e += 256) {
      int row = e >> 7, col = e & 127;
      int tg = row0 + row;
      float f = rfac[row];
      float val   = CT[row][col] * f;
      float other = CT[row][col ^ 64] * f;   // RoPE partner (col +- 64)
      int j = col & 63;
      float ang = (j < 32) ? exp2f(-10.0f * (float)j * (1.0f / 31.0f)) : 0.0f;
      float th = (float)tg * ang;
      float cth = cosf(th), sth = sinf(th);
      float o = (col < 64) ? (val * cth + other * sth) : (val * cth - other * sth);
      dst[(size_t)tg * DMODEL + n0 + col] = (bf16_t)o;
    }
  }
}

// ---------------- flash attention: one head x 128 queries per block ----------------
__global__ void __launch_bounds__(256) attn_kernel(const bf16_t* __restrict__ qb,
                                                   const bf16_t* __restrict__ kb,
                                                   const bf16_t* __restrict__ vb,
                                                   bf16_t* __restrict__ ob) {
  __shared__ bf16_t Kt[64][128];     // keys (row=key, col=d)        16 KB
  __shared__ bf16_t Vt[128][64];     // values transposed [d][key]   16 KB
  __shared__ bf16_t Pt[8][16][64];   // per-wave P staging           16 KB

  const int tid = threadIdx.x, lane = tid & 31, w = tid >> 5;
  const int l16 = lane & 15, hi = lane >> 4;
  const int q0 = blockIdx.x * 128;
  const int h  = blockIdx.y;

  // Q A-fragments kept in registers for whole loop (wave owns 16 query rows)
  v16bf qf[4];
  {
    const bf16_t* qrow = qb + (size_t)(q0 + 16 * w + l16) * DMODEL + h * HDHEAD;
#pragma unroll
    for (int kk = 0; kk < 4; ++kk) {
      const bf16_t* p = qrow + 32 * kk + hi * 8;
#pragma unroll
      for (int i = 0; i < 8; ++i) { qf[kk][i] = p[i]; qf[kk][i + 8] = p[i + 16]; }
    }
  }

  v8f oacc[8];
#pragma unroll
  for (int nt = 0; nt < 8; ++nt) oacc[nt] = (v8f){0.f,0.f,0.f,0.f,0.f,0.f,0.f,0.f};
  float mrun[8], lrun[8];
#pragma unroll
  for (int r = 0; r < 8; ++r) { mrun[r] = -3.0e38f; lrun[r] = 0.f; }
  const float scale = 0.08838834764831845f;  // 1/sqrt(128)

  for (int j0 = 0; j0 < q0 + 128; j0 += 64) {   // causal: only key blocks <= query block
    __syncthreads();
    // K tile via Tensor Data Mover (row-major, contiguous B-frags for S = Q*K^T)
#if HAVE_TDM
    if (w == 0)
      tdm_load_bf16_2d(&Kt[0][0], kb + (size_t)j0 * DMODEL + h * HDHEAD, 128, 64, DMODEL);
#else
#pragma unroll
    for (int it = 0; it < 4; ++it) {
      int c = tid + it * 256;
      int key = c >> 4, dc = (c & 15) * 8;
      *(uint4*)&Kt[key][dc] = *(const uint4*)(kb + (size_t)(j0 + key) * DMODEL + h * HDHEAD + dc);
    }
#endif
    // V tile transposed (contiguous B-frags for O += P*V)
#pragma unroll
    for (int it = 0; it < 32; ++it) {
      int e = tid + it * 256;
      int key = e >> 7, d = e & 127;
      Vt[d][key] = vb[(size_t)(j0 + key) * DMODEL + h * HDHEAD + d];
    }
    // warm GL2 for the next key block
    if (j0 + 64 < q0 + 128) {
      __builtin_prefetch(kb + (size_t)(j0 + 64 + (tid >> 2)) * DMODEL + h * HDHEAD, 0, 1);
      __builtin_prefetch(vb + (size_t)(j0 + 64 + (tid >> 2)) * DMODEL + h * HDHEAD, 0, 1);
    }
#if HAVE_TDM
    if (w == 0) __builtin_amdgcn_s_wait_tensorcnt(0);   // DMA visible before barrier
#endif
    __syncthreads();

    // S = Q * K^T  (16 wmma per wave)
    v8f sacc[4];
#pragma unroll
    for (int t = 0; t < 4; ++t) sacc[t] = (v8f){0.f,0.f,0.f,0.f,0.f,0.f,0.f,0.f};
#pragma unroll
    for (int t = 0; t < 4; ++t)
#pragma unroll
      for (int kk = 0; kk < 4; ++kk) {
        v16bf b = load_bfrag(&Kt[16 * t][0], 128, 32 * kk, lane);
        sacc[t] = wmma_bf16(qf[kk], b, sacc[t]);
      }

    // scale + causal mask
    const int qrg = q0 + 16 * w + hi * 8;
#pragma unroll
    for (int t = 0; t < 4; ++t) {
      int kc = j0 + 16 * t + l16;
#pragma unroll
      for (int r = 0; r < 8; ++r) {
        float s = sacc[t][r] * scale;
        sacc[t][r] = (kc > qrg + r) ? -1.0e30f : s;
      }
    }

    // online softmax
    float mnew[8], alpha[8];
#pragma unroll
    for (int r = 0; r < 8; ++r) {
      float v = fmaxf(fmaxf(sacc[0][r], sacc[1][r]), fmaxf(sacc[2][r], sacc[3][r]));
#pragma unroll
      for (int m = 8; m >= 1; m >>= 1) v = fmaxf(v, __shfl_xor(v, m, 16));
      mnew[r] = fmaxf(mrun[r], v);
      alpha[r] = __expf(mrun[r] - mnew[r]);
      mrun[r] = mnew[r];
    }
#pragma unroll
    for (int t = 0; t < 4; ++t)
#pragma unroll
      for (int r = 0; r < 8; ++r)
        sacc[t][r] = __expf(sacc[t][r] - mnew[r]);
#pragma unroll
    for (int r = 0; r < 8; ++r) {
      float s = sacc[0][r] + sacc[1][r] + sacc[2][r] + sacc[3][r];
#pragma unroll
      for (int m = 8; m >= 1; m >>= 1) s += __shfl_xor(s, m, 16);
      lrun[r] = lrun[r] * alpha[r] + s;
    }
#pragma unroll
    for (int nt = 0; nt < 8; ++nt)
#pragma unroll
      for (int r = 0; r < 8; ++r)
        oacc[nt][r] *= alpha[r];

    // P: C layout -> A layout via wave-private LDS
#pragma unroll
    for (int t = 0; t < 4; ++t)
#pragma unroll
      for (int r = 0; r < 8; ++r)
        Pt[w][r + hi * 8][16 * t + l16] = (bf16_t)sacc[t][r];

    // O += P * V  (16 wmma per wave)
#pragma unroll
    for (int kk = 0; kk < 2; ++kk) {
      v16bf pa = load_afrag(&Pt[w][0][0], 64, 32 * kk, lane);
#pragma unroll
      for (int nt = 0; nt < 8; ++nt) {
        v16bf b = load_bfrag(&Vt[16 * nt][0], 64, 32 * kk, lane);
        oacc[nt] = wmma_bf16(pa, b, oacc[nt]);
      }
    }
  }

  // finalize: divide by softmax denominator, write bf16
#pragma unroll
  for (int nt = 0; nt < 8; ++nt)
#pragma unroll
    for (int r = 0; r < 8; ++r) {
      float o = oacc[nt][r] / lrun[r];
      ob[(size_t)(q0 + 16 * w + r + hi * 8) * DMODEL + h * HDHEAD + 16 * nt + l16] = (bf16_t)o;
    }
}

// ---------------- output projection ----------------
__global__ void __launch_bounds__(256) proj_kernel(const bf16_t* __restrict__ ob,
                                                   const float* __restrict__ proj_w,
                                                   float* __restrict__ out) {
  __shared__ bf16_t At[64][64];
  __shared__ bf16_t Wt[128][64];
  const int tid = threadIdx.x, lane = tid & 31, w = tid >> 5;
  const int l16 = lane & 15, hi = lane >> 4;
  const int row0 = blockIdx.x * 64;
  const int n0   = blockIdx.y * 128;

  v8f acc[4];
  gemm_tile(ob, DMODEL, proj_w, DMODEL, row0, n0, DMODEL, At, Wt, acc);

  const int rw = (w & 3) * 16, cw = (w >> 2) * 64;
#pragma unroll
  for (int t = 0; t < 4; ++t)
#pragma unroll
    for (int r = 0; r < 8; ++r)
      out[(size_t)(row0 + rw + r + hi * 8) * DMODEL + n0 + cw + 16 * t + l16] = acc[t][r];
}

extern "C" void kernel_launch(void* const* d_in, const int* in_sizes, int n_in,
                              void* d_out, int out_size, void* d_ws, size_t ws_size,
                              hipStream_t stream) {
  const float* x       = (const float*)d_in[0];
  const float* ve      = (const float*)d_in[1];
  const float* lambdas = (const float*)d_in[2];
  const float* qkv_w   = (const float*)d_in[3];
  const float* proj_w  = (const float*)d_in[4];
  float* out = (float*)d_out;

  const size_t NE = (size_t)T_SEQ * DMODEL;   // 8M elements
  bf16_t* xb = (bf16_t*)d_ws;                 // 16 MB each
  bf16_t* qb = xb + NE;
  bf16_t* kb = qb + NE;
  bf16_t* vb = kb + NE;
  bf16_t* ob = vb + NE;

  cvt_kernel <<<dim3((unsigned)(NE / (256 * 4))), dim3(256), 0, stream>>>(x, xb);
  qkv_kernel <<<dim3(T_SEQ / 64, NHEAD, 3),      dim3(256), 0, stream>>>(xb, qkv_w, ve, lambdas, qb, kb, vb);
  attn_kernel<<<dim3(T_SEQ / 128, NHEAD),        dim3(256), 0, stream>>>(qb, kb, vb, ob);
  proj_kernel<<<dim3(T_SEQ / 64, DMODEL / 128),  dim3(256), 0, stream>>>(ob, proj_w, out);
}
